// Lanczos_DRC_block_38809324486715
// MI455X (gfx1250) — matrix operand order repaired and checked by reading, not compiled, not run on hardware
//
#include <hip/hip_runtime.h>
#include <hip/hip_bf16.h>

// ---------------------------------------------------------------------------
// Lanczos DRC block, factorized through rank-K spectral form.
//   P = Qr^T real + Qi^T imag            [K,C]
//   M = Qi^T real - Qr^T imag            [K,C]
//   PWs = TT[:,None]*(P @ W), MWs = TT[:,None]*(M @ W)   (TT = Ritz^long_diff)
//   res_real = Qr@PWs + Qi@MWs ; res_imag = Qi@PWs - Qr@MWs
//   out = (real + mask*res_real, imag + mask*res_imag), mask = res_real>=0
// All big GEMMs use v_wmma_f32_16x16x32_f16 (f16 in, f32 accumulate).
// ---------------------------------------------------------------------------

#define N_ROWS 10000
#define KDIM   64
#define CDIM   128
#define P1_BLOCKS 40          // split-K partials: 40 blocks x 256 rows = 10240 >= N
#define PM_ELEMS (2 * KDIM * CDIM)   // P followed by M, 16384 floats

typedef _Float16 v16h __attribute__((ext_vector_type(16)));
typedef _Float16 h8   __attribute__((ext_vector_type(8)));
typedef float    v8f  __attribute__((ext_vector_type(8)));

__device__ __forceinline__ v8f wmma_f16(v16h a, v16h b, v8f c) {
  // (neg_a, A, neg_b, B, c_mod, C, reuse_a, reuse_b)
  return __builtin_amdgcn_wmma_f32_16x16x32_f16(false, a, false, b, (short)0, c,
                                                false, false);
}

// 16 halves from two 8-half groups (16B-aligned LDS) -> one fragment.
__device__ __forceinline__ v16h frag2(const _Float16* p0, const _Float16* p1) {
  h8 lo = *(const h8*)p0;
  h8 hi = *(const h8*)p1;
  return __builtin_shufflevector(lo, hi, 0, 1, 2, 3, 4, 5, 6, 7,
                                 8, 9, 10, 11, 12, 13, 14, 15);
}

// Fragment built from two groups of 8 consecutive f32 in global memory.
__device__ __forceinline__ v16h fragg(const float* p0, const float* p1) {
  v16h a;
#pragma unroll
  for (int j = 0; j < 8; ++j) {
    a[j]     = (_Float16)p0[j];
    a[8 + j] = (_Float16)p1[j];
  }
  return a;
}

__device__ __forceinline__ v16h negf(v16h x) {
  v16h r;
#pragma unroll
  for (int i = 0; i < 16; ++i) r[i] = -x[i];
  return r;
}

// ---------------------------------------------------------------------------
// Phase 1: per-block partial P,M over a 256-row slice of N (deterministic
// split-K, no float atomics). Output rows = k (64), cols = c (128),
// inner = n chunks of 32 (the WMMA K).
// ---------------------------------------------------------------------------
__global__ __launch_bounds__(256) void p1_partial(const float* __restrict__ real,
                                                  const float* __restrict__ imag,
                                                  const float* __restrict__ Qr,
                                                  const float* __restrict__ Qi,
                                                  float* __restrict__ partial) {
  __shared__ __align__(16) _Float16 sA[2][KDIM][32];  // [Qr/Qi][k][n]
  __shared__ __align__(16) _Float16 sB[2][CDIM][32];  // [real/imag][c][n]

  const int tid  = threadIdx.x;
  const int wave = tid >> 5;        // 0..7 -> c-tile
  const int lane = tid & 31;
  const int lm   = lane & 15;
  const int hi16 = lane >> 4;       // 0/1
  const int b8   = hi16 ? 8 : 0;    // A-fragment n-offset
  const int b16  = hi16 ? 16 : 0;   // unused; B fragments also use b8

  const v8f zero8 = {0.f, 0.f, 0.f, 0.f, 0.f, 0.f, 0.f, 0.f};
  v8f accP[4], accM[4];
#pragma unroll
  for (int t = 0; t < 4; ++t) { accP[t] = zero8; accM[t] = zero8; }

  const int nbase0 = blockIdx.x * 256;
  for (int ch = 0; ch < 8; ++ch) {
    const int nb = nbase0 + ch * 32;
    __syncthreads();
    // Stage A side: Qr^T, Qi^T as [k][n] f16 (zero-pad rows >= N).
    for (int idx = tid; idx < 2 * KDIM * 32; idx += 256) {
      const int which = idx >> 11;        // 0=Qr, 1=Qi
      const int rem   = idx & 2047;
      const int n = rem >> 6, k = rem & 63;
      const int row = nb + n;
      float v = 0.f;
      if (row < N_ROWS) v = (which ? Qi : Qr)[row * KDIM + k];
      sA[which][k][n] = (_Float16)v;
    }
    // Stage B side: real, imag as [c][n] f16.
    for (int idx = tid; idx < 2 * CDIM * 32; idx += 256) {
      const int which = idx >> 12;        // 0=real, 1=imag
      const int rem   = idx & 4095;
      const int n = rem >> 7, c = rem & 127;
      const int row = nb + n;
      float v = 0.f;
      if (row < N_ROWS) v = (which ? imag : real)[row * CDIM + c];
      sB[which][c][n] = (_Float16)v;
    }
    __syncthreads();

    const int c0 = wave * 16 + lm;
    const v16h bR  = frag2(&sB[0][c0][b8], &sB[0][c0][16 + b8]);
    const v16h bI  = frag2(&sB[1][c0][b8], &sB[1][c0][16 + b8]);
    const v16h bIn = negf(bI);
#pragma unroll
    for (int kt = 0; kt < 4; ++kt) {
      const int k0 = kt * 16 + lm;
      const v16h aR = frag2(&sA[0][k0][b8], &sA[0][k0][16 + b8]);
      const v16h aI = frag2(&sA[1][k0][b8], &sA[1][k0][16 + b8]);
      accP[kt] = wmma_f16(aR, bR, accP[kt]);   // P += Qr^T real
      accP[kt] = wmma_f16(aI, bI, accP[kt]);   // P += Qi^T imag
      accM[kt] = wmma_f16(aI, bR, accM[kt]);   // M += Qi^T real
      accM[kt] = wmma_f16(aR, bIn, accM[kt]);  // M -= Qr^T imag
    }
  }

  float* pp = partial + (size_t)blockIdx.x * PM_ELEMS;
#pragma unroll
  for (int kt = 0; kt < 4; ++kt) {
#pragma unroll
    for (int v = 0; v < 8; ++v) {
      const int k = kt * 16 + v + b8;
      const int c = wave * 16 + lm;
      pp[k * CDIM + c]               = accP[kt][v];
      pp[KDIM * CDIM + k * CDIM + c] = accM[kt][v];
    }
  }
}

// Deterministic serial reduction of the 40 partial slabs -> P,M.
__global__ void p1_reduce(const float* __restrict__ partial,
                          float* __restrict__ PM) {
  const int i = blockIdx.x * blockDim.x + threadIdx.x;
  if (i < PM_ELEMS) {
    float s = 0.f;
    for (int b = 0; b < P1_BLOCKS; ++b) s += partial[(size_t)b * PM_ELEMS + i];
    PM[i] = s;
  }
}

// ---------------------------------------------------------------------------
// Phase 2 (single block): PW = P@W, MW = M@W, then scale row k by Ritz^ld.
// Output rows = k (64), cols = c2 (128), inner = c1 (128).
// ---------------------------------------------------------------------------
__global__ __launch_bounds__(256) void p2_small(const float* __restrict__ PM,
                                                const float* __restrict__ W,
                                                const float* __restrict__ Ritz,
                                                const int* __restrict__ ldp,
                                                float* __restrict__ PMW) {
  __shared__ __align__(16) _Float16 sW[CDIM][CDIM];  // W^T: [c2][c1]

  const int tid  = threadIdx.x;
  const int wave = tid >> 5;
  const int lane = tid & 31;
  const int lm   = lane & 15;
  const int hi16 = lane >> 4;
  const int b8   = hi16 ? 8 : 0;
  const int b16  = hi16 ? 16 : 0;

  for (int idx = tid; idx < CDIM * CDIM; idx += 256) {
    const int c2 = idx & 127, c1 = idx >> 7;
    sW[c2][c1] = (_Float16)W[c1 * CDIM + c2];
  }
  __syncthreads();

  const float* P = PM;
  const float* M = PM + KDIM * CDIM;
  const v8f zero8 = {0.f, 0.f, 0.f, 0.f, 0.f, 0.f, 0.f, 0.f};
  v8f accP[4], accM[4];
#pragma unroll
  for (int t = 0; t < 4; ++t) { accP[t] = zero8; accM[t] = zero8; }

  const int c2 = wave * 16 + lm;
#pragma unroll
  for (int ch = 0; ch < 4; ++ch) {
    const int n0 = ch * 32;
    const v16h bW = frag2(&sW[c2][n0 + b8], &sW[c2][n0 + 16 + b8]);
#pragma unroll
    for (int kt = 0; kt < 4; ++kt) {
      const int k = kt * 16 + lm;
      const v16h aP = fragg(&P[k * CDIM + n0 + b8], &P[k * CDIM + n0 + 16 + b8]);
      const v16h aM = fragg(&M[k * CDIM + n0 + b8], &M[k * CDIM + n0 + 16 + b8]);
      accP[kt] = wmma_f16(aP, bW, accP[kt]);
      accM[kt] = wmma_f16(aM, bW, accM[kt]);
    }
  }

  const int ld = ldp[0];
#pragma unroll
  for (int kt = 0; kt < 4; ++kt) {
#pragma unroll
    for (int v = 0; v < 8; ++v) {
      const int k = kt * 16 + v + b8;
      const float tt = powf(Ritz[k], (float)ld);
      PMW[k * CDIM + c2]               = accP[kt][v] * tt;
      PMW[KDIM * CDIM + k * CDIM + c2] = accM[kt][v] * tt;
    }
  }
}

// ---------------------------------------------------------------------------
// Phase 3: res_real = Qr@PWs + Qi@MWs, res_imag = Qi@PWs - Qr@MWs, fused
// complex-ReLU residual epilogue. 125 blocks x 5 slabs of 16 rows.
// ---------------------------------------------------------------------------
__global__ __launch_bounds__(256) void p3_apply(const float* __restrict__ real,
                                                const float* __restrict__ imag,
                                                const float* __restrict__ Qr,
                                                const float* __restrict__ Qi,
                                                const float* __restrict__ PMW,
                                                float* __restrict__ out) {
  __shared__ __align__(16) _Float16 sP[CDIM][KDIM];  // PWs^T: [c][k]
  __shared__ __align__(16) _Float16 sM[CDIM][KDIM];  // MWs^T: [c][k]

  const int tid  = threadIdx.x;
  const int wave = tid >> 5;
  const int lane = tid & 31;
  const int lm   = lane & 15;
  const int hi16 = lane >> 4;
  const int b8   = hi16 ? 8 : 0;
  const int b16  = hi16 ? 16 : 0;

  for (int idx = tid; idx < CDIM * KDIM; idx += 256) {
    const int c = idx & 127, k = idx >> 7;
    sP[c][k] = (_Float16)PMW[k * CDIM + c];
    sM[c][k] = (_Float16)PMW[KDIM * CDIM + k * CDIM + c];
  }
  __syncthreads();

  const int c0 = wave * 16 + lm;
  const v16h bP0  = frag2(&sP[c0][b8],      &sP[c0][16 + b8]);
  const v16h bP1  = frag2(&sP[c0][32 + b8], &sP[c0][48 + b8]);
  const v16h bM0  = frag2(&sM[c0][b8],      &sM[c0][16 + b8]);
  const v16h bM1  = frag2(&sM[c0][32 + b8], &sM[c0][48 + b8]);
  const v16h bM0n = negf(bM0);
  const v16h bM1n = negf(bM1);
  const v8f zero8 = {0.f, 0.f, 0.f, 0.f, 0.f, 0.f, 0.f, 0.f};

  for (int s = 0; s < 5; ++s) {
    const int row0 = (blockIdx.x * 5 + s) * 16;
    const int r = row0 + lm;                     // A-matrix M index = lane%16
    const v16h aR0 = fragg(&Qr[r * KDIM + b8],      &Qr[r * KDIM + 16 + b8]);
    const v16h aR1 = fragg(&Qr[r * KDIM + 32 + b8], &Qr[r * KDIM + 48 + b8]);
    const v16h aI0 = fragg(&Qi[r * KDIM + b8],      &Qi[r * KDIM + 16 + b8]);
    const v16h aI1 = fragg(&Qi[r * KDIM + 32 + b8], &Qi[r * KDIM + 48 + b8]);

    v8f cR = zero8, cI = zero8;
    cR = wmma_f16(aR0, bP0, cR);
    cR = wmma_f16(aR1, bP1, cR);
    cR = wmma_f16(aI0, bM0, cR);
    cR = wmma_f16(aI1, bM1, cR);
    cI = wmma_f16(aI0, bP0, cI);
    cI = wmma_f16(aI1, bP1, cI);
    cI = wmma_f16(aR0, bM0n, cI);
    cI = wmma_f16(aR1, bM1n, cI);

#pragma unroll
    for (int v = 0; v < 8; ++v) {
      const int rr = row0 + v + b8;              // C/D row = v + 8*hi16
      const int c  = c0;
      const float rv = real[rr * CDIM + c];
      const float iv = imag[rr * CDIM + c];
      const float resR = cR[v];
      const float resI = cI[v];
      const float m = (resR >= 0.f) ? 1.f : 0.f;
      out[rr * CDIM + c]                   = rv + m * resR;
      out[N_ROWS * CDIM + rr * CDIM + c]   = iv + m * resI;
    }
  }
}

// ---------------------------------------------------------------------------
extern "C" void kernel_launch(void* const* d_in, const int* in_sizes, int n_in,
                              void* d_out, int out_size, void* d_ws, size_t ws_size,
                              hipStream_t stream) {
  (void)in_sizes; (void)n_in; (void)out_size; (void)ws_size;
  const float* real  = (const float*)d_in[0];
  const float* imag  = (const float*)d_in[1];
  const float* Qr    = (const float*)d_in[2];
  const float* Qi    = (const float*)d_in[3];
  const float* Ritz  = (const float*)d_in[4];
  const float* W     = (const float*)d_in[5];
  const int*   ldp   = (const int*)d_in[6];
  float* out = (float*)d_out;

  // Workspace layout (floats):
  //   [0, 16384)               P then M
  //   [16384, 32768)           PWs then MWs
  //   [32768, 32768+40*16384)  split-K partials (~2.6 MB total)
  float* ws      = (float*)d_ws;
  float* PM      = ws;
  float* PMW     = ws + PM_ELEMS;
  float* partial = ws + 2 * PM_ELEMS;

  p1_partial<<<P1_BLOCKS, 256, 0, stream>>>(real, imag, Qr, Qi, partial);
  p1_reduce<<<(PM_ELEMS + 255) / 256, 256, 0, stream>>>(partial, PM);
  p2_small<<<1, 256, 0, stream>>>(PM, W, Ritz, ldp, PMW);
  p3_apply<<<125, 256, 0, stream>>>(real, imag, Qr, Qi, PMW, out);
}
